// GATParser_10471130267777
// MI455X (gfx1250) — matrix operand order repaired and checked by reading, not compiled, not run on hardware
//
#include <hip/hip_runtime.h>
#include <hip/hip_bf16.h>

// ============================================================================
// GAT dependency parser for MI455X (gfx1250, wave32, WMMA).
// All dense GEMMs (~250 GFLOP) run through one tiled kernel built around
// v_wmma_f32_16x16x32_bf16 (f32 accumulate): 256 threads = 8 waves, block
// tile 128(M)x64(N), each wave owns a 32x32 output block (2x2 WMMA tiles ->
// 4 wmma issues per K-step with A/B fragment reuse). Tiles are staged through
// LDS as bf16 via float4 global loads + packed cvt. Irregular ops (softmax,
// top-k, GATv2 edge softmax + scatter) are lightweight VALU/atomic kernels.
// ============================================================================

// ---------------- problem constants ----------------
constexpr int Bc      = 32;
constexpr int S0c     = 511;
constexpr int Sc      = 512;          // S0 + sentinel
constexpr int Ec      = 768;
constexpr int ARCc    = 500;
constexpr int TAGc    = 100;
constexpr int Hc      = 2;
constexpr int TOPKc   = 3;
constexpr int LAYERSc = 2;
constexpr int BSc     = Bc * Sc;      // 16384 total rows
constexpr int ECNTc   = Sc * TOPKc;   // 1536 edges per sample
constexpr int LD1c    = 504;          // padded leading dim for (ARC+1)-wide buffers
constexpr float NEG_SLOPEc = 0.2f;

typedef __attribute__((ext_vector_type(16))) __bf16 v16bf;
typedef __attribute__((ext_vector_type(8)))  float  v8f;

static inline int cdiv(int a, int b) { return (a + b - 1) / b; }

__device__ inline unsigned pack_bf16(float a, float b) {
    unsigned short ul = __builtin_bit_cast(unsigned short, (__bf16)a);
    unsigned short uh = __builtin_bit_cast(unsigned short, (__bf16)b);
    return (unsigned)ul | ((unsigned)uh << 16);
}

// ============================================================================
// Tiled WMMA GEMM:  C[b] = act(alpha * (A[b] @ op(B[b])) + bias)
//   A: [M,K] row-major (lda); op(B): [K,N]; transB=1 means B stored [N,K].
//   bias_mode: 0 none, 1 per-column vector, 2 scalar.  act: 0 none, 1 ELU.
// ============================================================================
#define TM 128
#define TN 64
#define TK 32
#define LDSTR 40   // TK + 8 pad: 80B row stride -> 8B aligned, conflict-spread

__device__ inline void store_tile(float* __restrict__ C, int ldc, int M, int N,
                                  int rbase, int cbase, const v8f& acc,
                                  const float* __restrict__ bias, int bias_mode,
                                  float alpha, int act, int lmL, int hiL)
{
    int col = cbase + lmL;
    if (col >= N) return;
    float bv = 0.f;
    if (bias_mode == 1)      bv = bias[col];
    else if (bias_mode == 2) bv = bias[0];
    #pragma unroll
    for (int r = 0; r < 8; ++r) {
        int row = rbase + r + hiL * 8;
        if (row < M) {
            float v = alpha * acc[r] + bv;
            if (act == 1) v = (v > 0.f) ? v : (__expf(v) - 1.f);
            C[(long long)row * ldc + col] = v;
        }
    }
}

__global__ __launch_bounds__(256)
void wmma_gemm(const float* __restrict__ A, long long sAb, int lda,
               const float* __restrict__ Bm, long long sBb, int ldb, int transB,
               const float* __restrict__ bias, int bias_mode,
               float* __restrict__ C, long long sCb, int ldc,
               int M, int N, int K, float alpha, int act)
{
    __shared__ __bf16 sA[TM * LDSTR];
    __shared__ __bf16 sB[TN * LDSTR];

    const int bz = blockIdx.z;
    A  += (long long)bz * sAb;
    Bm += (long long)bz * sBb;
    C  += (long long)bz * sCb;

    const int tid  = threadIdx.x;
    const int lane = tid & 31;
    const int wave = tid >> 5;
    const int wm   = wave >> 1;          // 0..3 : 32-row band
    const int wn   = wave & 1;           // 0..1 : 32-col band

    const int m0 = blockIdx.y * TM;
    const int n0 = blockIdx.x * TN;

    const int hiL = lane >> 4;           // lane half (ISA fragment layout)
    const int lmL = lane & 15;

    const bool fullM = (m0 + TM <= M);
    const bool fullN = (n0 + TN <= N);
    const bool alnA  = ((lda & 3) == 0);
    const bool alnB  = ((ldb & 3) == 0);

    v8f acc00 = {}, acc01 = {}, acc10 = {}, acc11 = {};

    for (int k0 = 0; k0 < K; k0 += TK) {
        const bool fullK = (k0 + TK <= K);

        // ---------------- stage A tile (128 x 32) as bf16 ----------------
        if (fullM && fullK && alnA) {
            #pragma unroll
            for (int i = 0; i < 4; ++i) {
                int idx = tid + i * 256;               // 1024 float4 slots
                int r = idx >> 3, c4 = (idx & 7) * 4;
                const float4 v = *(const float4*)(A + (long long)(m0 + r) * lda + k0 + c4);
                uint2 p; p.x = pack_bf16(v.x, v.y); p.y = pack_bf16(v.z, v.w);
                *(uint2*)&sA[r * LDSTR + c4] = p;
            }
        } else {
            #pragma unroll
            for (int i = 0; i < 4; ++i) {
                int idx = tid + i * 256;
                int r = idx >> 3, c4 = (idx & 7) * 4;
                int gr = m0 + r;
                float x0 = 0.f, x1 = 0.f, x2 = 0.f, x3 = 0.f;
                if (gr < M) {
                    const float* ap = A + (long long)gr * lda + k0;
                    if (k0 + c4 + 0 < K) x0 = ap[c4 + 0];
                    if (k0 + c4 + 1 < K) x1 = ap[c4 + 1];
                    if (k0 + c4 + 2 < K) x2 = ap[c4 + 2];
                    if (k0 + c4 + 3 < K) x3 = ap[c4 + 3];
                }
                uint2 p; p.x = pack_bf16(x0, x1); p.y = pack_bf16(x2, x3);
                *(uint2*)&sA[r * LDSTR + c4] = p;
            }
        }

        // ---------------- stage B tile (64 x 32) into sB[n][k] ----------------
        if (transB) {
            if (fullN && fullK && alnB) {
                #pragma unroll
                for (int i = 0; i < 2; ++i) {
                    int idx = tid + i * 256;           // 512 float4 slots
                    int n = idx >> 3, c4 = (idx & 7) * 4;
                    const float4 v = *(const float4*)(Bm + (long long)(n0 + n) * ldb + k0 + c4);
                    uint2 p; p.x = pack_bf16(v.x, v.y); p.y = pack_bf16(v.z, v.w);
                    *(uint2*)&sB[n * LDSTR + c4] = p;
                }
            } else {
                #pragma unroll
                for (int i = 0; i < 8; ++i) {
                    int idx = tid + i * 256;           // 2048 elements
                    int n = idx >> 5, kk = idx & 31;
                    float v = 0.f;
                    int gn = n0 + n, gk = k0 + kk;
                    if (gn < N && gk < K) v = Bm[(long long)gn * ldb + gk];
                    sB[n * LDSTR + kk] = (__bf16)v;
                }
            }
        } else {
            if (fullN && fullK && alnB) {
                #pragma unroll
                for (int i = 0; i < 2; ++i) {
                    int idx = tid + i * 256;
                    int kk = idx >> 4, n4 = (idx & 15) * 4;
                    const float4 v = *(const float4*)(Bm + (long long)(k0 + kk) * ldb + n0 + n4);
                    sB[(n4 + 0) * LDSTR + kk] = (__bf16)v.x;
                    sB[(n4 + 1) * LDSTR + kk] = (__bf16)v.y;
                    sB[(n4 + 2) * LDSTR + kk] = (__bf16)v.z;
                    sB[(n4 + 3) * LDSTR + kk] = (__bf16)v.w;
                }
            } else {
                #pragma unroll
                for (int i = 0; i < 8; ++i) {
                    int idx = tid + i * 256;
                    int kk = idx >> 6, n = idx & 63;
                    float v = 0.f;
                    int gn = n0 + n, gk = k0 + kk;
                    if (gn < N && gk < K) v = Bm[(long long)gk * ldb + gn];
                    sB[n * LDSTR + kk] = (__bf16)v;
                }
            }
        }
        __syncthreads();

        // ---- fragments per ISA §7.12.2; 2 A rows x 2 B cols, 4 WMMAs ----
        v16bf a0, a1, b0, b1;
        const __bf16* ap0 = &sA[(wm * 32 + lmL) * LDSTR];
        const __bf16* ap1 = &sA[(wm * 32 + 16 + lmL) * LDSTR];
        const __bf16* bp0 = &sB[(wn * 32 + lmL) * LDSTR];
        const __bf16* bp1 = &sB[(wn * 32 + 16 + lmL) * LDSTR];
        #pragma unroll
        for (int vv = 0; vv < 8; ++vv) {
            int ka = ((vv >= 4) ? 16 : 0) + ((vv & 3) * 2) + hiL * 8;
            a0[2 * vv] = ap0[ka]; a0[2 * vv + 1] = ap0[ka + 1];
            a1[2 * vv] = ap1[ka]; a1[2 * vv + 1] = ap1[ka + 1];
            int kb = ((vv >= 4) ? 16 : 0) + ((vv & 3) * 2) + hiL * 8;
            b0[2 * vv] = bp0[kb]; b0[2 * vv + 1] = bp0[kb + 1];
            b1[2 * vv] = bp1[kb]; b1[2 * vv + 1] = bp1[kb + 1];
        }
        acc00 = __builtin_amdgcn_wmma_f32_16x16x32_bf16(false, a0, false, b0, (short)0, acc00, false, false);
        acc01 = __builtin_amdgcn_wmma_f32_16x16x32_bf16(false, a0, false, b1, (short)0, acc01, false, false);
        acc10 = __builtin_amdgcn_wmma_f32_16x16x32_bf16(false, a1, false, b0, (short)0, acc10, false, false);
        acc11 = __builtin_amdgcn_wmma_f32_16x16x32_bf16(false, a1, false, b1, (short)0, acc11, false, false);
        __syncthreads();
    }

    const int rb = m0 + wm * 32;
    const int cb = n0 + wn * 32;
    store_tile(C, ldc, M, N, rb,      cb,      acc00, bias, bias_mode, alpha, act, lmL, hiL);
    store_tile(C, ldc, M, N, rb,      cb + 16, acc01, bias, bias_mode, alpha, act, lmL, hiL);
    store_tile(C, ldc, M, N, rb + 16, cb,      acc10, bias, bias_mode, alpha, act, lmL, hiL);
    store_tile(C, ldc, M, N, rb + 16, cb + 16, acc11, bias, bias_mode, alpha, act, lmL, hiL);
}

// ============================================================================
// Small helper kernels
// ============================================================================
__global__ void fill_kernel(float* p, float v, long long n) {
    long long i = (long long)blockIdx.x * blockDim.x + threadIdx.x;
    if (i < n) p[i] = v;
}

__global__ void copy_kernel(const float* __restrict__ s, float* __restrict__ d, long long n) {
    long long i = (long long)blockIdx.x * blockDim.x + threadIdx.x;
    if (i < n) d[i] = s[i];
}

__global__ void compute_valid(const int* __restrict__ mask, float* valid, long long n) {
    long long i = (long long)blockIdx.x * blockDim.x + threadIdx.x;
    if (i < n) atomicAdd(valid, (float)mask[i]);
}

// build x = concat(sentinel, encoded) -> [B, S, E]
__global__ void build_x_kernel(const float* __restrict__ enc,
                               const float* __restrict__ sent,
                               float* __restrict__ X) {
    long long i = (long long)blockIdx.x * blockDim.x + threadIdx.x;
    const long long total = (long long)BSc * Ec;
    if (i >= total) return;
    int e = (int)(i % Ec);
    long long bs = i / Ec;
    int s = (int)(bs % Sc);
    int b = (int)(bs / Sc);
    X[i] = (s == 0) ? sent[e] : enc[((long long)b * S0c + (s - 1)) * Ec + e];
}

// src [BS, d] -> dst [BS, ldd]: col d = 1.0, cols (d, ldd) = 0
__global__ void append_ones_kernel(const float* __restrict__ src,
                                   float* __restrict__ dst, int d, int ldd) {
    long long i = (long long)blockIdx.x * blockDim.x + threadIdx.x;
    const long long total = (long long)BSc * ldd;
    if (i >= total) return;
    int c = (int)(i % ldd);
    long long r = i / ldd;
    dst[i] = (c < d) ? src[r * d + c] : ((c == d) ? 1.f : 0.f);
}

// pad [rows, scols] -> [rows, dcols] with zero fill
__global__ void pad_matrix_kernel(const float* __restrict__ src, float* __restrict__ dst,
                                  int rows, int scols, int dcols) {
    long long i = (long long)blockIdx.x * blockDim.x + threadIdx.x;
    const long long total = (long long)rows * dcols;
    if (i >= total) return;
    int c = (int)(i % dcols);
    long long r = i / dcols;
    dst[i] = (c < scols) ? src[r * scols + c] : 0.f;
}

// ============================================================================
// Row softmax (probs) + masked log-softmax loss contribution, one block/row.
// ============================================================================
__global__ __launch_bounds__(256)
void softmax_loss_kernel(const float* __restrict__ att,
                         const int* __restrict__ mask,
                         const int* __restrict__ head_indices,
                         float* __restrict__ probs,
                         float* __restrict__ sel_sum)
{
    __shared__ float red[256];
    const int row = blockIdx.x;                  // b*S + i
    const int b = row >> 9;
    const int i = row & (Sc - 1);
    const float* arow = att + (long long)row * Sc;
    float* prow = probs + (long long)row * Sc;
    const int tid = threadIdx.x;
    const int j0 = tid, j1 = tid + 256;

    float mf0 = (j0 == 0) ? 1.f : (float)mask[b * S0c + j0 - 1];
    float mf1 = (float)mask[b * S0c + j1 - 1];   // j1 >= 256
    float lg0 = __logf(mf0 + 1e-45f);
    float lg1 = __logf(mf1 + 1e-45f);
    float a0 = arow[j0], a1 = arow[j1];

    red[tid] = fmaxf(a0, a1); __syncthreads();
    for (int s = 128; s > 0; s >>= 1) { if (tid < s) red[tid] = fmaxf(red[tid], red[tid + s]); __syncthreads(); }
    float Pmax = red[0]; __syncthreads();

    red[tid] = fmaxf(a0 + lg0, a1 + lg1); __syncthreads();
    for (int s = 128; s > 0; s >>= 1) { if (tid < s) red[tid] = fmaxf(red[tid], red[tid + s]); __syncthreads(); }
    float Mmax = red[0]; __syncthreads();

    red[tid] = __expf(a0 - Pmax) + __expf(a1 - Pmax); __syncthreads();
    for (int s = 128; s > 0; s >>= 1) { if (tid < s) red[tid] += red[tid + s]; __syncthreads(); }
    float Psum = red[0]; __syncthreads();

    red[tid] = __expf(a0 + lg0 - Mmax) + __expf(a1 + lg1 - Mmax); __syncthreads();
    for (int s = 128; s > 0; s >>= 1) { if (tid < s) red[tid] += red[tid + s]; __syncthreads(); }
    float Msum = red[0];

    prow[j0] = __expf(a0 - Pmax) / Psum;
    prow[j1] = __expf(a1 - Pmax) / Psum;

    if (tid == 0 && i >= 1) {
        int hr = head_indices[b * S0c + i - 1];      // hi with sentinel prepended
        float mfh = (hr == 0) ? 1.f : (float)mask[b * S0c + hr - 1];
        float lgh = __logf(mfh + 1e-45f);
        float sel = (arow[hr] + lgh - Mmax - __logf(Msum)) * mfh;
        atomicAdd(sel_sum, sel);
    }
}

// ============================================================================
// top-3 over rows / columns of probs [B,S,S]
// ============================================================================
__device__ inline void top3_update(float v, int j, float& v0, float& v1, float& v2,
                                   int& i0, int& i1, int& i2) {
    if (v > v0)      { v2 = v1; i2 = i1; v1 = v0; i1 = i0; v0 = v; i0 = j; }
    else if (v > v1) { v2 = v1; i2 = i1; v1 = v;  i1 = j; }
    else if (v > v2) { v2 = v;  i2 = j; }
}

__global__ void topk_rows(const float* __restrict__ probs, float* __restrict__ ea, int* __restrict__ ei) {
    int r = blockIdx.x * blockDim.x + threadIdx.x;    // b*S + i
    if (r >= BSc) return;
    const float* p = probs + (long long)r * Sc;
    float v0 = -1e30f, v1 = -1e30f, v2 = -1e30f; int i0 = 0, i1 = 0, i2 = 0;
    for (int j = 0; j < Sc; ++j) top3_update(p[j], j, v0, v1, v2, i0, i1, i2);
    long long o = (long long)r * TOPKc;
    ea[o] = v0; ea[o + 1] = v1; ea[o + 2] = v2;
    ei[o] = i0; ei[o + 1] = i1; ei[o + 2] = i2;
}

__global__ void topk_cols(const float* __restrict__ probs, float* __restrict__ ea, int* __restrict__ ei) {
    int r = blockIdx.x * blockDim.x + threadIdx.x;    // b*S + j
    if (r >= BSc) return;
    int b = r >> 9, j = r & (Sc - 1);
    const float* p = probs + (long long)b * Sc * Sc + j;
    float v0 = -1e30f, v1 = -1e30f, v2 = -1e30f; int i0 = 0, i1 = 0, i2 = 0;
    for (int i = 0; i < Sc; ++i) top3_update(p[(long long)i * Sc], i, v0, v1, v2, i0, i1, i2);
    long long o = (long long)r * TOPKc;
    ea[o] = v0; ea[o + 1] = v1; ea[o + 2] = v2;
    ei[o] = i0; ei[o + 1] = i1; ei[o + 2] = i2;
}

// ============================================================================
// GATv2 edge kernels
// ============================================================================
__device__ inline void atomicMaxFloat(float* addr, float val) {
    unsigned int* ai = (unsigned int*)addr;
    unsigned int old = *ai;
    while (true) {
        float f = __uint_as_float(old);
        if (f >= val) break;
        unsigned int assumed = old;
        old = atomicCAS(ai, assumed, __float_as_uint(val));
        if (old == assumed) break;
    }
}

// one thread per (b, e, h): logit = sum_c leaky_relu(xl[src]+xr[tgt]+ea*We) . att
__global__ void gat_logit_kernel(const float* __restrict__ xl, const float* __restrict__ xr,
                                 const float* __restrict__ ea, const int* __restrict__ ei,
                                 const float* __restrict__ Wevec, const float* __restrict__ attw,
                                 float* __restrict__ logits, float* __restrict__ mx, int d)
{
    int idx = blockIdx.x * blockDim.x + threadIdx.x;
    const int total = Bc * ECNTc * Hc;
    if (idx >= total) return;
    int h = idx % Hc;
    int e = (idx / Hc) % ECNTc;
    int b = idx / (Hc * ECNTc);
    int src = e / TOPKc;
    int tgt = ei[b * ECNTc + e];
    float eattr = ea[b * ECNTc + e];
    const float* xls = xl + ((long long)(b * Sc + src) * Hc + h) * d;
    const float* xrt = xr + ((long long)(b * Sc + tgt) * Hc + h) * d;
    const float* we  = Wevec + h * d;
    const float* aw  = attw  + h * d;
    float acc = 0.f;
    for (int c = 0; c < d; ++c) {
        float m = xls[c] + xrt[c] + eattr * we[c];
        m = (m > 0.f) ? m : NEG_SLOPEc * m;
        acc += m * aw[c];
    }
    logits[idx] = acc;
    atomicMaxFloat(&mx[(b * Sc + tgt) * Hc + h], acc);
}

__global__ void gat_fix_mx(float* mx, long long n) {
    long long i = (long long)blockIdx.x * blockDim.x + threadIdx.x;
    if (i >= n) return;
    float v = mx[i];
    if (!(v > -1e30f)) mx[i] = 0.f;   // segment with no edges: -inf -> 0
}

__global__ void gat_alpha_kernel(const float* __restrict__ logits, const int* __restrict__ ei,
                                 const float* __restrict__ mx,
                                 float* __restrict__ alphaB, float* __restrict__ den)
{
    int idx = blockIdx.x * blockDim.x + threadIdx.x;
    const int total = Bc * ECNTc * Hc;
    if (idx >= total) return;
    int h = idx % Hc;
    int e = (idx / Hc) % ECNTc;
    int b = idx / (Hc * ECNTc);
    int tgt = ei[b * ECNTc + e];
    float a = __expf(logits[idx] - mx[(b * Sc + tgt) * Hc + h]);
    alphaB[idx] = a;
    atomicAdd(&den[(b * Sc + tgt) * Hc + h], a);
}

// one thread per (b, e, c): out[tgt,c] += mean_h( a_h/den_h * xl[src,h,c] )
__global__ void gat_aggregate(const float* __restrict__ alphaB, const float* __restrict__ den,
                              const int* __restrict__ ei, const float* __restrict__ xl,
                              float* __restrict__ out, int d)
{
    long long idx = (long long)blockIdx.x * blockDim.x + threadIdx.x;
    const long long total = (long long)Bc * ECNTc * d;
    if (idx >= total) return;
    int c = (int)(idx % d);
    int e = (int)((idx / d) % ECNTc);
    int b = (int)(idx / ((long long)d * ECNTc));
    int src = e / TOPKc;
    int tgt = ei[b * ECNTc + e];
    float s = 0.f;
    #pragma unroll
    for (int h = 0; h < Hc; ++h) {
        float a  = alphaB[(b * ECNTc + e) * Hc + h];
        float dn = den[(b * Sc + tgt) * Hc + h] + 1e-16f;
        s += (a / dn) * xl[((long long)(b * Sc + src) * Hc + h) * d + c];
    }
    atomicAdd(&out[((long long)(b * Sc + tgt)) * d + c], s * (1.f / Hc));
}

__global__ void finalize_losses(const float* __restrict__ scalars, float* __restrict__ out) {
    int l = threadIdx.x;
    if (l < LAYERSc) out[l] = -scalars[1 + l] / scalars[0];
}

// ============================================================================
// Host orchestration
// ============================================================================
static void launch_gemm(hipStream_t st,
                        const float* A, long long sAb, int lda,
                        const float* Bm, long long sBb, int ldb, int transB,
                        const float* bias, int bias_mode,
                        float* C, long long sCb, int ldc,
                        int M, int N, int K, float alpha, int act, int batch)
{
    dim3 grid(cdiv(N, TN), cdiv(M, TM), batch);
    wmma_gemm<<<grid, 256, 0, st>>>(A, sAb, lda, Bm, sBb, ldb, transB,
                                    bias, bias_mode, C, sCb, ldc, M, N, K, alpha, act);
}

static void run_gat_conv(hipStream_t st, const float* Xin, float* Xout, int d,
                         const float* Wl, const float* bl, const float* Wr, const float* br,
                         const float* attw, const float* Wevec, const float* Wres, const float* bvec,
                         const float* ea, const int* ei,
                         float* xl, float* xr, float* logits, float* alphaB,
                         float* mx, float* den)
{
    const int HD = Hc * d;
    launch_gemm(st, Xin, 0, d, Wl, 0, HD, 0, bl, 1, xl, 0, HD, BSc, HD, d, 1.0f, 0, 1);
    launch_gemm(st, Xin, 0, d, Wr, 0, HD, 0, br, 1, xr, 0, HD, BSc, HD, d, 1.0f, 0, 1);
    // residual + bias directly into the output buffer
    launch_gemm(st, Xin, 0, d, Wres, 0, d, 0, bvec, 1, Xout, 0, d, BSc, d, d, 1.0f, 0, 1);

    const long long nmx = (long long)Bc * Sc * Hc;
    fill_kernel<<<(unsigned)cdiv((int)nmx, 256), 256, 0, st>>>(mx, -1e38f, nmx);
    fill_kernel<<<(unsigned)cdiv((int)nmx, 256), 256, 0, st>>>(den, 0.f, nmx);

    const int nlog = Bc * ECNTc * Hc;
    gat_logit_kernel<<<cdiv(nlog, 256), 256, 0, st>>>(xl, xr, ea, ei, Wevec, attw, logits, mx, d);
    gat_fix_mx<<<(unsigned)cdiv((int)nmx, 256), 256, 0, st>>>(mx, nmx);
    gat_alpha_kernel<<<cdiv(nlog, 256), 256, 0, st>>>(logits, ei, mx, alphaB, den);

    const long long nag = (long long)Bc * ECNTc * d;
    gat_aggregate<<<(unsigned)((nag + 255) / 256), 256, 0, st>>>(alphaB, den, ei, xl, Xout, d);
}

extern "C" void kernel_launch(void* const* d_in, const int* in_sizes, int n_in,
                              void* d_out, int out_size, void* d_ws, size_t ws_size,
                              hipStream_t stream)
{
    (void)in_sizes; (void)n_in; (void)out_size; (void)ws_size;

    // ---- input pointer map (setup_inputs dict order, params flattened) ----
    const float* enc   = (const float*)d_in[0];
    const int*   mask  = (const int*)  d_in[2];
    const int*   hidx  = (const int*)  d_in[3];
    const float* sent  = (const float*)d_in[4];
    const float* ffW[4] = { (const float*)d_in[5], (const float*)d_in[7],
                            (const float*)d_in[9], (const float*)d_in[11] };
    const float* ffB[4] = { (const float*)d_in[6], (const float*)d_in[8],
                            (const float*)d_in[10], (const float*)d_in[12] };
    const float* bilW[3] = { (const float*)d_in[13], (const float*)d_in[15], (const float*)d_in[17] };
    const float* bilB[3] = { (const float*)d_in[14], (const float*)d_in[16], (const float*)d_in[18] };
    // GAT blocks: 8 tensors each (Wl,bl,Wr,br,att,We,Wres,b)
    auto gp = [&](int base, int l, int t) { return (const float*)d_in[base + l * 8 + t]; };
    const int G1A = 19, G2A = 35, G1R = 51, G2R = 67;

    // ---- workspace carve-out ----
    char* wsp = (char*)d_ws;
    auto allocF = [&](long long n) { float* p = (float*)wsp; wsp += ((n * 4 + 255) / 256) * 256; return p; };
    auto allocI = [&](long long n) { int*   p = (int*)wsp;   wsp += ((n * 4 + 255) / 256) * 256; return p; };

    float* X       = allocF((long long)BSc * Ec);
    float* harc[2] = { allocF((long long)BSc * ARCc), allocF((long long)BSc * ARCc) };
    float* darc[2] = { allocF((long long)BSc * ARCc), allocF((long long)BSc * ARCc) };
    float* htag[2] = { allocF((long long)BSc * TAGc), allocF((long long)BSc * TAGc) };
    float* dtag[2] = { allocF((long long)BSc * TAGc), allocF((long long)BSc * TAGc) };
    float* h1      = allocF((long long)BSc * LD1c);   // padded ld = 504
    float* d1      = allocF((long long)BSc * LD1c);
    float* tmp     = allocF((long long)BSc * LD1c);
    float* bilWp[3] = { allocF((long long)(ARCc + 1) * LD1c),
                        allocF((long long)(ARCc + 1) * LD1c),
                        allocF((long long)(ARCc + 1) * LD1c) };
    float* attbuf  = allocF((long long)Bc * Sc * Sc);
    float* probs   = allocF((long long)Bc * Sc * Sc);
    float* ea_r    = allocF((long long)Bc * ECNTc);
    int*   ei_r    = allocI((long long)Bc * ECNTc);
    float* ea_c    = allocF((long long)Bc * ECNTc);
    int*   ei_c    = allocI((long long)Bc * ECNTc);
    float* xl      = allocF((long long)BSc * Hc * ARCc);
    float* xr      = allocF((long long)BSc * Hc * ARCc);
    float* logits  = allocF((long long)Bc * ECNTc * Hc);
    float* alphaB  = allocF((long long)Bc * ECNTc * Hc);
    float* mx      = allocF((long long)Bc * Sc * Hc);
    float* den     = allocF((long long)Bc * Sc * Hc);
    float* scalars = allocF(8);   // [0]=valid, [1..2]=sel sums

    const float inv_sqrt_arc = 1.0f / sqrtf((float)ARCc);

    // ---- valid = sum(mask); zero loss accumulators ----
    fill_kernel<<<1, 32, 0, stream>>>(scalars, 0.f, 8);
    compute_valid<<<cdiv(Bc * S0c, 256), 256, 0, stream>>>(mask, &scalars[0], (long long)Bc * S0c);

    // ---- pad bilinear weights into 16B-aligned leading dim ----
    {
        long long n = (long long)(ARCc + 1) * LD1c;
        for (int l = 0; l < 3; ++l)
            pad_matrix_kernel<<<(unsigned)((n + 255) / 256), 256, 0, stream>>>(
                bilW[l], bilWp[l], ARCc + 1, ARCc + 1, LD1c);
    }

    // ---- x = concat(sentinel, encoded) ----
    {
        long long n = (long long)BSc * Ec;
        build_x_kernel<<<(unsigned)((n + 255) / 256), 256, 0, stream>>>(enc, sent, X);
    }

    // ---- FF projections with ELU (WMMA) ----
    launch_gemm(stream, X, 0, Ec, ffW[0], 0, ARCc, 0, ffB[0], 1, harc[0], 0, ARCc, BSc, ARCc, Ec, 1.0f, 1, 1);
    launch_gemm(stream, X, 0, Ec, ffW[1], 0, ARCc, 0, ffB[1], 1, darc[0], 0, ARCc, BSc, ARCc, Ec, 1.0f, 1, 1);
    launch_gemm(stream, X, 0, Ec, ffW[2], 0, TAGc, 0, ffB[2], 1, htag[0], 0, TAGc, BSc, TAGc, Ec, 1.0f, 1, 1);
    launch_gemm(stream, X, 0, Ec, ffW[3], 0, TAGc, 0, ffB[3], 1, dtag[0], 0, TAGc, BSc, TAGc, Ec, 1.0f, 1, 1);

    int cur = 0;
    const long long n_h1 = (long long)BSc * LD1c;

    for (int l = 0; l < LAYERSc; ++l) {
        // ---- bilinear attention: att = (h1 @ W) @ d1^T / sqrt(ARC) + b ----
        append_ones_kernel<<<(unsigned)((n_h1 + 255) / 256), 256, 0, stream>>>(harc[cur], h1, ARCc, LD1c);
        append_ones_kernel<<<(unsigned)((n_h1 + 255) / 256), 256, 0, stream>>>(darc[cur], d1, ARCc, LD1c);
        launch_gemm(stream, h1, 0, LD1c, bilWp[l], 0, LD1c, 0, nullptr, 0,
                    tmp, 0, LD1c, BSc, ARCc + 1, ARCc + 1, 1.0f, 0, 1);
        launch_gemm(stream, tmp, (long long)Sc * LD1c, LD1c,
                    d1, (long long)Sc * LD1c, LD1c, /*transB=*/1,
                    bilB[l], 2, attbuf, (long long)Sc * Sc, Sc,
                    Sc, Sc, ARCc + 1, inv_sqrt_arc, 0, Bc);

        // ---- softmax probs + masked log-softmax loss contribution ----
        softmax_loss_kernel<<<BSc, 256, 0, stream>>>(attbuf, mask, hidx, probs, &scalars[1 + l]);

        // ---- top-3 rows / cols -> edge lists ----
        topk_rows<<<cdiv(BSc, 256), 256, 0, stream>>>(probs, ea_r, ei_r);
        topk_cols<<<cdiv(BSc, 256), 256, 0, stream>>>(probs, ea_c, ei_c);

        // ---- four GATv2 convolutions ----
        run_gat_conv(stream, harc[cur], harc[cur ^ 1], ARCc,
                     gp(G1A, l, 0), gp(G1A, l, 1), gp(G1A, l, 2), gp(G1A, l, 3),
                     gp(G1A, l, 4), gp(G1A, l, 5), gp(G1A, l, 6), gp(G1A, l, 7),
                     ea_r, ei_r, xl, xr, logits, alphaB, mx, den);
        run_gat_conv(stream, darc[cur], darc[cur ^ 1], ARCc,
                     gp(G2A, l, 0), gp(G2A, l, 1), gp(G2A, l, 2), gp(G2A, l, 3),
                     gp(G2A, l, 4), gp(G2A, l, 5), gp(G2A, l, 6), gp(G2A, l, 7),
                     ea_c, ei_c, xl, xr, logits, alphaB, mx, den);
        run_gat_conv(stream, htag[cur], htag[cur ^ 1], TAGc,
                     gp(G1R, l, 0), gp(G1R, l, 1), gp(G1R, l, 2), gp(G1R, l, 3),
                     gp(G1R, l, 4), gp(G1R, l, 5), gp(G1R, l, 6), gp(G1R, l, 7),
                     ea_r, ei_r, xl, xr, logits, alphaB, mx, den);
        run_gat_conv(stream, dtag[cur], dtag[cur ^ 1], TAGc,
                     gp(G2R, l, 0), gp(G2R, l, 1), gp(G2R, l, 2), gp(G2R, l, 3),
                     gp(G2R, l, 4), gp(G2R, l, 5), gp(G2R, l, 6), gp(G2R, l, 7),
                     ea_c, ei_c, xl, xr, logits, alphaB, mx, den);
        cur ^= 1;
    }

    // ---- final bilinear -> attended (directly into d_out) ----
    float* out_att  = (float*)d_out;
    float* out_htag = out_att + (long long)Bc * Sc * Sc;
    float* out_dtag = out_htag + (long long)BSc * TAGc;
    float* out_loss = out_dtag + (long long)BSc * TAGc;

    append_ones_kernel<<<(unsigned)((n_h1 + 255) / 256), 256, 0, stream>>>(harc[cur], h1, ARCc, LD1c);
    append_ones_kernel<<<(unsigned)((n_h1 + 255) / 256), 256, 0, stream>>>(darc[cur], d1, ARCc, LD1c);
    launch_gemm(stream, h1, 0, LD1c, bilWp[2], 0, LD1c, 0, nullptr, 0,
                tmp, 0, LD1c, BSc, ARCc + 1, ARCc + 1, 1.0f, 0, 1);
    launch_gemm(stream, tmp, (long long)Sc * LD1c, LD1c,
                d1, (long long)Sc * LD1c, LD1c, 1,
                bilB[2], 2, out_att, (long long)Sc * Sc, Sc,
                Sc, Sc, ARCc + 1, inv_sqrt_arc, 0, Bc);

    // ---- copy final tags, finalize losses ----
    const long long ntag = (long long)BSc * TAGc;
    copy_kernel<<<(unsigned)((ntag + 255) / 256), 256, 0, stream>>>(htag[cur], out_htag, ntag);
    copy_kernel<<<(unsigned)((ntag + 255) / 256), 256, 0, stream>>>(dtag[cur], out_dtag, ntag);
    finalize_losses<<<1, 32, 0, stream>>>(scalars, out_loss);
}
